// BiLSTM_CRF_17978733101140
// MI455X (gfx1250) — matrix operand order repaired and checked by reading, not compiled, not run on hardware
//
#include <hip/hip_runtime.h>
#include <hip/hip_bf16.h>

typedef __attribute__((ext_vector_type(16))) _Float16 v16h;
typedef __attribute__((ext_vector_type(8)))  _Float16 v8h;
typedef __attribute__((ext_vector_type(8)))  float    v8f;

#define S_LEN 4096
#define E_DIM 256
#define H_DIM 256
#define G4H   1024   // 4*H
#define T_TAG 16
#define START_TAG 14
#define STOP_TAG  15
#define NEG_INF  -10000.0f

// ---------------------------------------------------------------------------
// Kernel 1: gather embeddings -> f16, convert weights -> f16, fuse biases.
// ---------------------------------------------------------------------------
__global__ __launch_bounds__(256) void prep_kernel(
    const int* __restrict__ sent, const float* __restrict__ embed,
    const float* __restrict__ wih_f, const float* __restrict__ wih_b,
    const float* __restrict__ whh_f, const float* __restrict__ whh_b,
    const float* __restrict__ bih_f, const float* __restrict__ bhh_f,
    const float* __restrict__ bih_b, const float* __restrict__ bhh_b,
    const float* __restrict__ w_out,
    _Float16* __restrict__ xh, _Float16* __restrict__ wih,
    _Float16* __restrict__ whh, _Float16* __restrict__ wout,
    float* __restrict__ bias)
{
    const int tid    = blockIdx.x * blockDim.x + threadIdx.x;
    const int stride = gridDim.x * blockDim.x;
    const int NX = S_LEN * E_DIM;     // 1,048,576
    const int NW = G4H * E_DIM;       // 262,144 per matrix

    for (int i = tid; i < NX; i += stride) {
        int s = i >> 8, e = i & 255;
        xh[i] = (_Float16)embed[(size_t)sent[s] * E_DIM + e];
    }
    for (int i = tid; i < NW; i += stride) {
        wih[i]      = (_Float16)wih_f[i];
        wih[NW + i] = (_Float16)wih_b[i];
        whh[i]      = (_Float16)whh_f[i];
        whh[NW + i] = (_Float16)whh_b[i];
    }
    for (int i = tid; i < T_TAG * 2 * H_DIM; i += stride)
        wout[i] = (_Float16)w_out[i];
    for (int i = tid; i < G4H; i += stride) {
        bias[i]       = bih_f[i] + bhh_f[i];
        bias[G4H + i] = bih_b[i] + bhh_b[i];
    }
}

// ---------------------------------------------------------------------------
// Kernel 2: xg[dir] = x @ W_ih[dir].T + (b_ih+b_hh)   via WMMA f16 -> f32.
// One wave computes a 16(M) x 64(N) strip: A fragment reused over 4 B tiles.
// ---------------------------------------------------------------------------
__global__ __launch_bounds__(256) void gemm_xg_kernel(
    const _Float16* __restrict__ xh, const _Float16* __restrict__ wih,
    const float* __restrict__ bias, float* __restrict__ xg)
{
    const int wave = (blockIdx.x * blockDim.x + threadIdx.x) >> 5; // 0..8191
    const int lane = threadIdx.x & 31;
    const int dir  = wave >> 12;          // 4096 wave-jobs per direction
    const int rem  = wave & 4095;
    const int m0   = (rem >> 4) * 16;     // 256 M tiles
    const int n0b  = (rem & 15) * 64;     // 16 groups of 4 N tiles

    const _Float16* W  = wih + (size_t)dir * (G4H * E_DIM);
    float* out         = xg  + (size_t)dir * ((size_t)S_LEN * G4H);

    const int l  = lane & 15;
    const int hi = lane >> 4;            // 0: K{0..15}  1: K{16..31}
    const _Float16* arow = xh + (size_t)(m0 + l) * E_DIM + hi * 16;

    v8f acc[4] = {};
    for (int k0 = 0; k0 < E_DIM; k0 += 32) {
        v8h alo = *(const v8h*)(arow + k0);
        v8h ahi = *(const v8h*)(arow + k0 + 8);
        v16h a  = __builtin_shufflevector(alo, ahi,
                    0,1,2,3,4,5,6,7,8,9,10,11,12,13,14,15);
#pragma unroll
        for (int j = 0; j < 4; ++j) {
            const int n0 = n0b + j * 16;
            // B[k][n] = W[n][k]; lane n holds a contiguous 16-half K slice.
            const _Float16* brow = W + (size_t)(n0 + l) * E_DIM + k0 + hi * 16;
            v16h b = *(const v16h*)brow;
            acc[j] = __builtin_amdgcn_wmma_f32_16x16x32_f16(
                         false, a, false, b, (short)0, acc[j], false, false);
        }
    }
    // D layout: lane holds column (lane&15), rows (lane>>4)*8 + r
    const int col = lane & 15, rbase = (lane >> 4) * 8;
#pragma unroll
    for (int j = 0; j < 4; ++j) {
        const int n = n0b + j * 16 + col;
        const float bn = bias[dir * G4H + n];
#pragma unroll
        for (int r = 0; r < 8; ++r)
            out[(size_t)(m0 + rbase + r) * G4H + n] = acc[j][r] + bn;
    }
}

// ---------------------------------------------------------------------------
// Kernel 3: sequential LSTM scan, one 1024-thread workgroup (one WGP) per
// direction. W_hh lives entirely in registers as WMMA B-fragments (wave w
// owns gate columns 32w..32w+31: 2 N-tiles x 8 K-frags = 128 VGPRs/thread).
// Per step: h (f16, LDS) is broadcast into a replicated A fragment (all 16
// rows = h; only D row 0 is consumed), 16 WMMAs/wave, gates combined via LDS.
// The next step's precomputed xg row is double-buffered into LDS with
// global_load_async_to_lds_b32 and retired with s_wait_asynccnt, hiding the
// global read under the WMMA phase.
// ---------------------------------------------------------------------------
__global__ __launch_bounds__(1024) void lstm_kernel(
    const float* __restrict__ xg, const _Float16* __restrict__ whh,
    const float* __restrict__ h0, const float* __restrict__ c0,
    _Float16* __restrict__ hsh)
{
    const int dir  = blockIdx.x;
    const int t    = threadIdx.x;       // 0..1023
    const int wv   = t >> 5;            // wave 0..31
    const int lane = t & 31;
    const int l    = lane & 15;
    const int hi   = lane >> 4;

    __shared__ __align__(16) _Float16 h16[H_DIM];
    __shared__ float gates[G4H];
    __shared__ __align__(16) float xg_buf[2][G4H];

    const float*    XG = xg  + (size_t)dir * ((size_t)S_LEN * G4H);
    const _Float16* W  = whh + (size_t)dir * (G4H * H_DIM);

    // ---- preload register-resident B fragments (W_hh^T tiles) ----
    v16h bfrag[2][8];
#pragma unroll
    for (int j = 0; j < 2; ++j) {
        const int n0 = (2 * wv + j) * 16;
        const _Float16* brow = W + (size_t)(n0 + l) * H_DIM + hi * 16;
#pragma unroll
        for (int kk = 0; kk < 8; ++kk)
            bfrag[j][kk] = *(const v16h*)(brow + kk * 32);
    }

    float c = 0.0f;
    if (t < H_DIM) { h16[t] = (_Float16)h0[dir * H_DIM + t]; c = c0[dir * H_DIM + t]; }

    // ---- async preload of xg row for step 0 ----
    {
        const int s0 = dir ? (S_LEN - 1) : 0;
        unsigned           loff = (unsigned)(size_t)&xg_buf[0][t];
        unsigned long long ga   = (unsigned long long)(size_t)&XG[(size_t)s0 * G4H + t];
        asm volatile("global_load_async_to_lds_b32 %0, %1, off"
                     :: "v"(loff), "v"(ga) : "memory");
        asm volatile("s_wait_asynccnt 0" ::: "memory");
    }
    __syncthreads();

    for (int step = 0; step < S_LEN; ++step) {
        const int s   = dir ? (S_LEN - 1 - step) : step;
        const int buf = step & 1;

        if (step + 1 < S_LEN) {                       // stage next xg row
            const int ns = dir ? (s - 1) : (s + 1);
            unsigned           loff = (unsigned)(size_t)&xg_buf[buf ^ 1][t];
            unsigned long long ga   = (unsigned long long)(size_t)&XG[(size_t)ns * G4H + t];
            asm volatile("global_load_async_to_lds_b32 %0, %1, off"
                         :: "v"(loff), "v"(ga) : "memory");
        }

        // ---- h @ W_hh^T via WMMA (row 0 of D is the matvec result) ----
        v8f acc0 = {}, acc1 = {};
#pragma unroll
        for (int kk = 0; kk < 8; ++kk) {
            const _Float16* hp = &h16[kk * 32 + hi * 16];
            v8h alo = *(const v8h*)(hp);
            v8h ahi = *(const v8h*)(hp + 8);
            v16h a  = __builtin_shufflevector(alo, ahi,
                        0,1,2,3,4,5,6,7,8,9,10,11,12,13,14,15);
            acc0 = __builtin_amdgcn_wmma_f32_16x16x32_f16(
                       false, a, false, bfrag[0][kk], (short)0, acc0, false, false);
            acc1 = __builtin_amdgcn_wmma_f32_16x16x32_f16(
                       false, a, false, bfrag[1][kk], (short)0, acc1, false, false);
        }

        // gate pre-activations: D row 0 = lanes 0..15, VGPR 0
        if (lane < 16) {
#pragma unroll
            for (int j = 0; j < 2; ++j) {
                const int col = (2 * wv + j) * 16 + l;
                float g = (j ? acc1[0] : acc0[0]) + xg_buf[buf][col];
                // PyTorch order i,f,g,o: rows [512,768) are tanh
                gates[col] = ((col >> 8) == 2) ? tanhf(g)
                                               : 1.0f / (1.0f + __expf(-g));
            }
        }
        __syncthreads();

        if (t < H_DIM) {
            const float i  = gates[t];
            const float f  = gates[H_DIM + t];
            const float gg = gates[2 * H_DIM + t];
            const float o  = gates[3 * H_DIM + t];
            c = f * c + i * gg;
            const float h  = o * tanhf(c);
            const _Float16 hh = (_Float16)h;
            h16[t] = hh;
            hsh[(size_t)s * (2 * H_DIM) + dir * H_DIM + t] = hh;
        }
        if (step + 1 < S_LEN)
            asm volatile("s_wait_asynccnt 0" ::: "memory");
        __syncthreads();
    }
}

// ---------------------------------------------------------------------------
// Kernel 4: feats = [hf|hb] @ W_out.T + b_out  via WMMA (N = 16 = one tile).
// ---------------------------------------------------------------------------
__global__ __launch_bounds__(256) void feats_kernel(
    const _Float16* __restrict__ hsh, const _Float16* __restrict__ wout,
    const float* __restrict__ bout, float* __restrict__ feats)
{
    const int wave = (blockIdx.x * blockDim.x + threadIdx.x) >> 5; // 0..255
    const int lane = threadIdx.x & 31;
    const int m0   = wave * 16;
    const int l    = lane & 15;
    const int hi   = lane >> 4;
    const int K    = 2 * H_DIM;

    const _Float16* arow = hsh + (size_t)(m0 + l) * K + hi * 16;
    v8f acc = {};
    for (int k0 = 0; k0 < K; k0 += 32) {
        v8h alo = *(const v8h*)(arow + k0);
        v8h ahi = *(const v8h*)(arow + k0 + 8);
        v16h a  = __builtin_shufflevector(alo, ahi,
                    0,1,2,3,4,5,6,7,8,9,10,11,12,13,14,15);
        const _Float16* brow = wout + (size_t)l * K + k0 + hi * 16;
        v16h b = *(const v16h*)brow;
        acc = __builtin_amdgcn_wmma_f32_16x16x32_f16(
                  false, a, false, b, (short)0, acc, false, false);
    }
    const int col = lane & 15, rbase = (lane >> 4) * 8;
    const float bn = bout[col];
#pragma unroll
    for (int r = 0; r < 8; ++r)
        feats[(size_t)(m0 + rbase + r) * T_TAG + col] = acc[r] + bn;
}

// ---------------------------------------------------------------------------
// Kernel 5: CRF forward algorithm, single wave32. Lane i owns tag row i.
// ---------------------------------------------------------------------------
__global__ __launch_bounds__(32) void crf_kernel(
    const float* __restrict__ feats, const float* __restrict__ trans,
    float* __restrict__ out)
{
    __shared__ float alpha[T_TAG];
    __shared__ float red[T_TAG];
    const int lane = threadIdx.x;

    float trow[T_TAG];
    if (lane < T_TAG) {
#pragma unroll
        for (int j = 0; j < T_TAG; ++j) trow[j] = trans[lane * T_TAG + j];
        alpha[lane] = (lane == START_TAG) ? 0.0f : NEG_INF;
    }
    __syncthreads();

    for (int s = 0; s < S_LEN; ++s) {
        float a_new = 0.0f;
        if (lane < T_TAG) {
            float v[T_TAG];
#pragma unroll
            for (int j = 0; j < T_TAG; ++j) v[j] = alpha[j] + trow[j];
            float m = v[0];
#pragma unroll
            for (int j = 1; j < T_TAG; ++j) m = fmaxf(m, v[j]);
            float sum = 0.0f;
#pragma unroll
            for (int j = 0; j < T_TAG; ++j) sum += __expf(v[j] - m);
            a_new = m + __logf(sum) + feats[s * T_TAG + lane];
        }
        __syncthreads();
        if (lane < T_TAG) alpha[lane] = a_new;
        __syncthreads();
    }

    if (lane < T_TAG) red[lane] = alpha[lane] + trans[STOP_TAG * T_TAG + lane];
    __syncthreads();
    if (lane == 0) {
        float m = red[0];
#pragma unroll
        for (int j = 1; j < T_TAG; ++j) m = fmaxf(m, red[j]);
        float sum = 0.0f;
#pragma unroll
        for (int j = 0; j < T_TAG; ++j) sum += __expf(red[j] - m);
        out[0] = m + __logf(sum);
    }
}

// ---------------------------------------------------------------------------
extern "C" void kernel_launch(void* const* d_in, const int* in_sizes, int n_in,
                              void* d_out, int out_size, void* d_ws, size_t ws_size,
                              hipStream_t stream) {
    const int*   sent  = (const int*)  d_in[0];
    const float* embed = (const float*)d_in[1];
    const float* wih_f = (const float*)d_in[2];
    const float* whh_f = (const float*)d_in[3];
    const float* bih_f = (const float*)d_in[4];
    const float* bhh_f = (const float*)d_in[5];
    const float* wih_b = (const float*)d_in[6];
    const float* whh_b = (const float*)d_in[7];
    const float* bih_b = (const float*)d_in[8];
    const float* bhh_b = (const float*)d_in[9];
    const float* w_out = (const float*)d_in[10];
    const float* b_out = (const float*)d_in[11];
    const float* trans = (const float*)d_in[12];
    const float* h0    = (const float*)d_in[13];
    const float* c0    = (const float*)d_in[14];
    float*       out   = (float*)d_out;

    // Workspace layout (bytes, all regions >= 32B aligned)
    char* ws = (char*)d_ws;
    size_t off = 0;
    _Float16* xh    = (_Float16*)(ws + off); off += (size_t)S_LEN * E_DIM * 2;       // 2 MB
    _Float16* wih16 = (_Float16*)(ws + off); off += (size_t)2 * G4H * E_DIM * 2;     // 1 MB
    _Float16* whh16 = (_Float16*)(ws + off); off += (size_t)2 * G4H * H_DIM * 2;     // 1 MB
    _Float16* wo16  = (_Float16*)(ws + off); off += (size_t)T_TAG * 2 * H_DIM * 2;   // 16 KB
    float*    bias  = (float*)   (ws + off); off += (size_t)2 * G4H * 4;             // 8 KB
    float*    xg    = (float*)   (ws + off); off += (size_t)2 * S_LEN * G4H * 4;     // 32 MB
    _Float16* hsh   = (_Float16*)(ws + off); off += (size_t)S_LEN * 2 * H_DIM * 2;   // 4 MB
    float*    feats = (float*)   (ws + off); off += (size_t)S_LEN * T_TAG * 4;       // 256 KB

    prep_kernel<<<1024, 256, 0, stream>>>(
        sent, embed, wih_f, wih_b, whh_f, whh_b,
        bih_f, bhh_f, bih_b, bhh_b, w_out,
        xh, wih16, whh16, wo16, bias);

    gemm_xg_kernel<<<1024, 256, 0, stream>>>(xh, wih16, bias, xg);

    lstm_kernel<<<2, 1024, 0, stream>>>(xg, whh16, h0, c0, hsh);

    feats_kernel<<<32, 256, 0, stream>>>(hsh, wo16, b_out, feats);

    crf_kernel<<<1, 32, 0, stream>>>(feats, trans, out);
}
